// GATDecoder_35115652612278
// MI455X (gfx1250) — matrix operand order repaired and checked by reading, not compiled, not run on hardware
//
#include <hip/hip_runtime.h>
#include <hip/hip_bf16.h>
#include <math.h>

typedef __attribute__((ext_vector_type(16))) _Float16 v16h;
typedef __attribute__((ext_vector_type(8)))  float    v8f;
typedef __attribute__((ext_vector_type(4)))  float    v4f;

#define NNODES 50000
#define NEDGES 800000
#define SLOPE  0.2f
#define TPB    256

// ---------- order-preserving float<->uint key for atomic max ----------
__device__ __forceinline__ unsigned f32_key(float f) {
  unsigned u = __float_as_uint(f);
  return (u & 0x80000000u) ? ~u : (u | 0x80000000u);
}
__device__ __forceinline__ float key_f32(unsigned k) {
  unsigned u = (k & 0x80000000u) ? (k ^ 0x80000000u) : ~k;
  return __uint_as_float(u);
}
#define NEGINF_KEY 0x007FFFFFu  // f32_key(-inf)

// ---------- conversion kernels ----------
__global__ void k_f32_to_f16(const float* __restrict__ in, _Float16* __restrict__ out, int n) {
  int i = blockIdx.x * blockDim.x + threadIdx.x;
  if (i < n) out[i] = (_Float16)in[i];
}

// W: [K, HC] row-major (f32)  ->  wT: [HC, K] row-major (f16)
__global__ void k_w_transpose(const float* __restrict__ W, _Float16* __restrict__ wT,
                              int K, int HC) {
  int i = blockIdx.x * blockDim.x + threadIdx.x;
  if (i >= K * HC) return;
  int j = i / K;        // output row = column of W
  int k = i - j * K;    // along K
  wT[i] = (_Float16)W[(size_t)k * HC + j];
}

// ---------- WMMA GEMM, 1x4 N-blocked + double-buffered K pipeline -------------
// Each wave computes a 16x64 slab of C. A/B fragments for step k+1 are loaded
// into their own registers before step k's WMMAs consume their operands, so
// the 10 b128 loads of the next step overlap the 4 WMMAs of the current one.
union HVec { v16h h; v4f f[2]; };

__global__ void k_wmma_gemm(const _Float16* __restrict__ A,
                            const _Float16* __restrict__ BT,
                            float* __restrict__ C,
                            int M, int K, int Nout, int total_waves) {
  int wave = (blockIdx.x * blockDim.x + threadIdx.x) >> 5;
  if (wave >= total_waves) return;          // wave-uniform guard
  int lane = threadIdx.x & 31;
  int ntn4 = Nout >> 6;                     // groups of 4 N-tiles (64 cols)
  int tm   = wave / ntn4;
  int tn4  = wave - tm * ntn4;

  int half = lane >> 4;                     // K-half select (ISA 16-bit layouts)
  int l16  = lane & 15;
  const _Float16* arow = A + (size_t)(tm * 16 + l16) * K;
  const _Float16* brow[4];
  #pragma unroll
  for (int t = 0; t < 4; ++t)
    brow[t] = BT + (size_t)(tn4 * 64 + t * 16 + l16) * K;

  v8f  acc[4] = {};
  HVec a[2];
  HVec b[2][4];

  // Preload K-step 0.
  // A 16x32 f16 layout: lanes 0-15 hold K0..7 & K16..23; lanes 16-31 K8..15 & K24..31
  a[0].f[0] = *(const v4f*)(arow + half * 8);
  a[0].f[1] = *(const v4f*)(arow + 16 + half * 8);
  #pragma unroll
  for (int t = 0; t < 4; ++t) {
    // B 32x16 f16 layout: lanes 0-15 K0..7 & K16..23, lanes 16-31 K8..15 & K24..31 (col N = lane&15)
    b[0][t].f[0] = *(const v4f*)(brow[t] + half * 8);
    b[0][t].f[1] = *(const v4f*)(brow[t] + 16 + half * 8);
  }

  #pragma unroll
  for (int kk = 0; kk < 8; ++kk) {
    int cur = kk & 1, nxt = cur ^ 1;
    if (kk < 7) {
      int kn = (kk + 1) * 32;
      a[nxt].f[0] = *(const v4f*)(arow + kn + half * 8);
      a[nxt].f[1] = *(const v4f*)(arow + kn + 16 + half * 8);
      #pragma unroll
      for (int t = 0; t < 4; ++t) {
        b[nxt][t].f[0] = *(const v4f*)(brow[t] + kn + half * 8);
        b[nxt][t].f[1] = *(const v4f*)(brow[t] + kn + 16 + half * 8);
      }
    }
    #pragma unroll
    for (int t = 0; t < 4; ++t)
      acc[t] = __builtin_amdgcn_wmma_f32_16x16x32_f16(false, a[cur].h, false, b[cur][t].h,
                                                      (short)0, acc[t], false, false);
  }

  // C/D layout: lanes 0-15 -> N=lane, M=r ; lanes 16-31 -> N=lane-16, M=8+r
  int row0 = tm * 16 + half * 8;
  #pragma unroll
  for (int t = 0; t < 4; ++t) {
    int col = tn4 * 64 + t * 16 + l16;
    #pragma unroll
    for (int r = 0; r < 8; ++r)
      C[(size_t)(row0 + r) * Nout + col] = acc[t][r];
  }
}

// ---------- a_s / a_d dots (v4f vectorized; all offsets 16B aligned) ----------
__global__ void k_attn_dots(const float* __restrict__ h,
                            const float* __restrict__ att_src,
                            const float* __restrict__ att_dst,
                            float* __restrict__ a_s, float* __restrict__ a_d,
                            int n, int H, int Cc) {
  int i = blockIdx.x * blockDim.x + threadIdx.x;
  if (i >= n * H) return;
  int node = i / H;
  int hh   = i - node * H;
  const v4f* hp = (const v4f*)(h + (size_t)node * H * Cc + hh * Cc);
  const v4f* sp = (const v4f*)(att_src + hh * Cc);
  const v4f* dp = (const v4f*)(att_dst + hh * Cc);
  float s = 0.f, d = 0.f;
  int c4 = Cc >> 2;
  for (int c = 0; c < c4; ++c) {
    v4f v = hp[c], sv = sp[c], dv = dp[c];
    s += v.x * sv.x + v.y * sv.y + v.z * sv.z + v.w * sv.w;
    d += v.x * dv.x + v.y * dv.y + v.z * dv.z + v.w * dv.w;
  }
  a_s[i] = s; a_d[i] = d;
}

// ---------- init: out = bias, m = -inf key, denom = 0 ----------
__global__ void k_init(float* __restrict__ out, const float* __restrict__ b,
                       unsigned* __restrict__ mkey, float* __restrict__ denom,
                       int n, int H, int HC) {
  int i = blockIdx.x * blockDim.x + threadIdx.x;
  if (i >= n * HC) return;
  int j = i & (HC - 1);                     // HC is a power of two
  out[i] = b[j];
  if (i < n * H) { mkey[i] = NEGINF_KEY; denom[i] = 0.f; }
}

// ---------- edge pass 1: logits + segment max ----------
__global__ void k_edge_logit_max(const int* __restrict__ src, const int* __restrict__ dst,
                                 const float* __restrict__ a_s, const float* __restrict__ a_d,
                                 float* __restrict__ ebuf, unsigned* __restrict__ mkey,
                                 int H, int lgH) {
  int i = blockIdx.x * blockDim.x + threadIdx.x;
  if (i >= NEDGES * H) return;
  int e  = i >> lgH;
  int hh = i & (H - 1);
  int s = src[e], d = dst[e];
  float v = a_s[(s << lgH) + hh] + a_d[(d << lgH) + hh];
  v = v > 0.f ? v : SLOPE * v;              // leaky_relu
  ebuf[i] = v;
  atomicMax(&mkey[(d << lgH) + hh], f32_key(v));
}

// ---------- edge pass 2: exp + segment sum ----------
__global__ void k_edge_exp_sum(const int* __restrict__ dst,
                               float* __restrict__ ebuf,
                               const unsigned* __restrict__ mkey,
                               float* __restrict__ denom, int H, int lgH) {
  int i = blockIdx.x * blockDim.x + threadIdx.x;
  if (i >= NEDGES * H) return;
  int e  = i >> lgH;
  int hh = i & (H - 1);
  int d = dst[e];
  float m = key_f32(mkey[(d << lgH) + hh]);
  if (!isfinite(m)) m = 0.f;                // jnp.where(isfinite(m), m, 0)
  float v = __expf(ebuf[i] - m);
  ebuf[i] = v;
  atomicAdd(&denom[(d << lgH) + hh], v);
}

// ---------- edge pass 3: alpha = e / (denom[dst] + eps) ----------
__global__ void k_edge_alpha(const int* __restrict__ dst,
                             float* __restrict__ ebuf,
                             const float* __restrict__ denom, int H, int lgH) {
  int i = blockIdx.x * blockDim.x + threadIdx.x;
  if (i >= NEDGES * H) return;
  int e  = i >> lgH;
  int hh = i & (H - 1);
  ebuf[i] = ebuf[i] / (denom[(dst[e] << lgH) + hh] + 1e-16f);
}

// ---------- edge pass 4: scatter out[dst] += alpha * h[src] --------------------
// Block of 256 threads covers (256/hc4) edges; all index math is shifts/masks.
__global__ void k_scatter(const int* __restrict__ src, const int* __restrict__ dst,
                          const float* __restrict__ alpha,
                          const float* __restrict__ h,
                          float* __restrict__ out,
                          int HC, int lg_hc4, int lgC, int lgH) {
  int t    = threadIdx.x;
  int eloc = t >> lg_hc4;                       // edge within block
  int r    = t & ((1 << lg_hc4) - 1);           // float4 slot within row
  int e    = (blockIdx.x << (8 - lg_hc4)) + eloc;
  int j    = r << 2;                            // channel offset
  int head = j >> lgC;
  float a = alpha[(e << lgH) + head];
  int s = src[e], d = dst[e];
  v4f hv = *(const v4f*)(h + (size_t)s * HC + j);
  float* op = out + (size_t)d * HC + j;
  atomicAdd(op + 0, a * hv.x);
  atomicAdd(op + 1, a * hv.y);
  atomicAdd(op + 2, a * hv.z);
  atomicAdd(op + 3, a * hv.w);
}

// ---------------------------------------------------------------------------
extern "C" void kernel_launch(void* const* d_in, const int* in_sizes, int n_in,
                              void* d_out, int out_size, void* d_ws, size_t ws_size,
                              hipStream_t stream) {
  (void)in_sizes; (void)n_in; (void)out_size; (void)ws_size;
  const float* x   = (const float*)d_in[0];
  const int*   ei  = (const int*)d_in[1];
  const int*   src = ei;
  const int*   dst = ei + NEDGES;

  struct Layer { const float *W, *as, *ad, *b; int H, C, lgH, lgC; };
  Layer L[3] = {
    { (const float*)d_in[3],  (const float*)d_in[4],  (const float*)d_in[5],  (const float*)d_in[6],  4,  64, 2, 6 },
    { (const float*)d_in[7],  (const float*)d_in[8],  (const float*)d_in[9],  (const float*)d_in[10], 4,  64, 2, 6 },
    { (const float*)d_in[11], (const float*)d_in[12], (const float*)d_in[13], (const float*)d_in[14], 1, 128, 0, 7 },
  };

  char* ws = (char*)d_ws;
  size_t off = 0;
  auto salloc = [&](size_t bytes) -> void* {
    void* p = ws + off;
    off += (bytes + 255) & ~(size_t)255;
    return p;
  };
  float*     hbuf  = (float*)    salloc((size_t)NNODES * 256 * 4);
  float*     ping  = (float*)    salloc((size_t)NNODES * 256 * 4);
  _Float16*  xf16  = (_Float16*) salloc((size_t)NNODES * 256 * 2);
  _Float16*  wT    = (_Float16*) salloc((size_t)256 * 256 * 2);
  float*     a_s   = (float*)    salloc((size_t)NNODES * 4 * 4);
  float*     a_d   = (float*)    salloc((size_t)NNODES * 4 * 4);
  unsigned*  mkey  = (unsigned*) salloc((size_t)NNODES * 4 * 4);
  float*     denom = (float*)    salloc((size_t)NNODES * 4 * 4);
  float*     ebuf  = (float*)    salloc((size_t)NEDGES * 4 * 4);

  const float* cur = x;
  const int K = 256;
  for (int l = 0; l < 3; ++l) {
    int H = L[l].H, Cc = L[l].C, HC = H * Cc;
    int lgH = L[l].lgH, lgC = L[l].lgC;
    float* outbuf = (l == 2) ? (float*)d_out : ping;   // input ping is dead after GEMM

    int ncvt = NNODES * K;
    k_f32_to_f16<<<(ncvt + TPB - 1) / TPB, TPB, 0, stream>>>(cur, xf16, ncvt);
    k_w_transpose<<<(HC * K + TPB - 1) / TPB, TPB, 0, stream>>>(L[l].W, wT, K, HC);

    int waves = (NNODES / 16) * (HC / 64);             // 1x4 N-blocked tiles
    k_wmma_gemm<<<(waves * 32 + TPB - 1) / TPB, TPB, 0, stream>>>(xf16, wT, hbuf,
                                                                  NNODES, K, HC, waves);

    k_attn_dots<<<(NNODES * H + TPB - 1) / TPB, TPB, 0, stream>>>(hbuf, L[l].as, L[l].ad,
                                                                  a_s, a_d, NNODES, H, Cc);
    k_init<<<(NNODES * HC + TPB - 1) / TPB, TPB, 0, stream>>>(outbuf, L[l].b, mkey, denom,
                                                              NNODES, H, HC);

    int eh = NEDGES * H;
    k_edge_logit_max<<<(eh + TPB - 1) / TPB, TPB, 0, stream>>>(src, dst, a_s, a_d, ebuf,
                                                               mkey, H, lgH);
    k_edge_exp_sum  <<<(eh + TPB - 1) / TPB, TPB, 0, stream>>>(dst, ebuf, mkey, denom, H, lgH);
    k_edge_alpha    <<<(eh + TPB - 1) / TPB, TPB, 0, stream>>>(dst, ebuf, denom, H, lgH);

    int hc4 = HC / 4;                                   // 64 or 32 float4 slots per row
    int lg_hc4 = (hc4 == 64) ? 6 : 5;
    int edges_per_block = TPB / hc4;                    // 4 or 8
    k_scatter<<<NEDGES / edges_per_block, TPB, 0, stream>>>(src, dst, ebuf, hbuf, outbuf,
                                                            HC, lg_hc4, lgC, lgH);
    cur = outbuf;
  }
}
